// TransformerUnit_56100862821058
// MI455X (gfx1250) — compile-verified, matrix operand order checked by reading, not run on hardware
//
#include <hip/hip_runtime.h>
#include <hip/hip_fp16.h>
#include <math.h>

typedef __attribute__((ext_vector_type(16))) _Float16 v16h;
typedef __attribute__((ext_vector_type(8)))  float    v8f;

#define N_B   16
#define T_LEN 300
#define V_J   25
#define C_CH  16
#define D_M   400      // C*V
#define H_N   2
#define DH    200
#define FF_M  1600
#define L_W   9
#define W_CNT 291      // T - L
#define NWIN  (N_B * W_CNT)   // 4656
#define NROWS (N_B * T_LEN)   // 4800
#define KS_D  13       // ceil(400/32)

// ---------------- WMMA fragment helpers (16x16x32 f16 -> f32) ----------------

__device__ __forceinline__ int frag_k(int g, int half) {
  return ((g < 4) ? (g * 2) : (16 + (g - 4) * 2)) + half * 8;
}

__device__ __forceinline__ v8f wmma_acc(v16h a, v16h b, v8f c) {
  return __builtin_amdgcn_wmma_f32_16x16x32_f16(false, a, false, b, (short)0, c,
                                                false, false);
}

// A fragment from row-major f32 source (LDS or global). Rows >= Mvalid or k >= K read 0.
__device__ __forceinline__ v16h load_a_f32(const float* A, int lda, int Mvalid,
                                           int kbase, int K, int lane) {
  int m = lane & 15, half = lane >> 4;
  v16h r;
#pragma unroll
  for (int g = 0; g < 8; ++g) {
    int k = kbase + frag_k(g, half);
    float x0 = 0.f, x1 = 0.f;
    if (m < Mvalid && k < K) {
      const float* p = A + m * lda + k;
      x0 = p[0]; x1 = p[1];
    }
    r[2 * g]     = (_Float16)x0;
    r[2 * g + 1] = (_Float16)x1;
  }
  return r;
}

// A fragment from row-major f16 source (LDS or global).
__device__ __forceinline__ v16h load_a_f16(const _Float16* A, int lda, int Mvalid,
                                           int kbase, int K, int lane) {
  int m = lane & 15, half = lane >> 4;
  v16h r;
#pragma unroll
  for (int g = 0; g < 8; ++g) {
    int k = kbase + frag_k(g, half);
    _Float16 x0 = (_Float16)0.f, x1 = (_Float16)0.f;
    if (m < Mvalid && k < K) {
      const _Float16* p = A + m * lda + k;
      x0 = p[0]; x1 = p[1];
    }
    r[2 * g]     = x0;
    r[2 * g + 1] = x1;
  }
  return r;
}

// B fragment: computes A @ W^T where W is [Ntot, K] row-major f16.
// B[k][n] = W[n][k]; lane holds N = n0 + lane%16, contiguous half2 along K.
__device__ __forceinline__ v16h load_b_w(const _Float16* W, int K, int n0, int kbase,
                                         int lane) {
  int n = n0 + (lane & 15), half = lane >> 4;
  v16h r;
#pragma unroll
  for (int g = 0; g < 8; ++g) {
    int k = kbase + frag_k(g, half);
    _Float16 x0 = (_Float16)0.f, x1 = (_Float16)0.f;
    if (k < K) {
      const _Float16* p = W + n * K + k;
      x0 = p[0]; x1 = p[1];
    }
    r[2 * g]     = x0;
    r[2 * g + 1] = x1;
  }
  return r;
}

// K-loop with register-resident A fragments and 1-deep software-pipelined B loads:
// loads for step ks+1 are issued before the WMMA of step ks.
template <int KS>
__device__ __forceinline__ v8f gemm_pipelined(const v16h* a, const _Float16* W, int K,
                                              int n0, int lane) {
  v8f acc = {};
  v16h bc = load_b_w(W, K, n0, 0, lane);
#pragma unroll
  for (int ks = 0; ks < KS; ++ks) {
    v16h bn = bc;
    if (ks + 1 < KS) bn = load_b_w(W, K, n0, (ks + 1) * 32, lane);
    acc = wmma_acc(a[ks], bc, acc);
    bc = bn;
  }
  return acc;
}

// ---------------- Kernel 1: layout change + positional encoding ----------------
// xs[n][t][d] = x[n][c][t][v] + pe[t][d],  d = v*16 + c  (torch view order)
__global__ void prep_xs_kernel(const float* __restrict__ x, const float* __restrict__ pe,
                               float* __restrict__ xs, _Float16* __restrict__ xs_h) {
  int idx = blockIdx.x * blockDim.x + threadIdx.x;
  if (idx >= NROWS * D_M) return;
  int n = idx / (T_LEN * D_M);
  int r = idx % (T_LEN * D_M);
  int t = r / D_M;
  int d = r % D_M;
  int v = d >> 4;
  int c = d & 15;
  float val = x[((n * C_CH + c) * T_LEN + t) * V_J + v] + pe[t * D_M + d];
  xs[idx]   = val;
  xs_h[idx] = (_Float16)val;
}

// ---------------- Kernel 2: f32 -> f16 weight conversion ----------------
__global__ void cvt_f16_kernel(const float* __restrict__ src, _Float16* __restrict__ dst,
                               int count) {
  int idx = blockIdx.x * blockDim.x + threadIdx.x;
  if (idx < count) dst[idx] = (_Float16)src[idx];
}

// ---------------- Kernel 3: hoisted QKV GEMM over all 4800 rows ----------------
// qkv_all[4800][1200] = xs[4800][400] @ w_in^T + b_in
// One block per 16-row M-tile; 8 waves cover the 75 N-tiles; A fragments hoisted
// into registers once per wave (13 x v16h = 104 VGPRs) and reused across tiles.
__global__ void __launch_bounds__(256, 1)
qkv_gemm_kernel(const _Float16* __restrict__ xs_h,
                const _Float16* __restrict__ w_in_h,
                const float* __restrict__ b_in,
                float* __restrict__ qkv_all) {
  int wv = threadIdx.x >> 5, lane = threadIdx.x & 31;
  int mt = blockIdx.x;  // 0..299
  const _Float16* Arow = xs_h + (size_t)mt * 16 * D_M;

  v16h a13[KS_D];
#pragma unroll
  for (int ks = 0; ks < KS_D; ++ks)
    a13[ks] = load_a_f16(Arow, D_M, 16, ks * 32, D_M, lane);

  for (int nt = wv; nt < 1200 / 16; nt += 8) {   // wave-uniform
    v8f acc = gemm_pipelined<KS_D>(a13, w_in_h, D_M, nt * 16, lane);
    int half = lane >> 4, nn = nt * 16 + (lane & 15);
    float bias = b_in[nn];
#pragma unroll
    for (int g = 0; g < 8; ++g) {
      int m = mt * 16 + g + half * 8;
      qkv_all[m * 1200 + nn] = acc[g] + bias;
    }
  }
}

// ---------------- wave-parallel LayerNorm over 9 rows of width 400 ----------------
__device__ __forceinline__ void layernorm_rows(float* h, const float* g, const float* b,
                                               int wv, int lane) {
  for (int r = wv; r < L_W; r += 8) {
    float s = 0.f;
    for (int c = lane; c < D_M; c += 32) s += h[r * D_M + c];
#pragma unroll
    for (int off = 16; off > 0; off >>= 1) s += __shfl_xor(s, off, 32);
    float mu = s * (1.0f / D_M);
    float vv = 0.f;
    for (int c = lane; c < D_M; c += 32) {
      float t = h[r * D_M + c] - mu;
      vv += t * t;
    }
#pragma unroll
    for (int off = 16; off > 0; off >>= 1) vv += __shfl_xor(vv, off, 32);
    float rs = rsqrtf(vv * (1.0f / D_M) + 1e-5f);
    for (int c = lane; c < D_M; c += 32)
      h[r * D_M + c] = (h[r * D_M + c] - mu) * rs * g[c] + b[c];
  }
}

// ---------------- Kernel 4: one workgroup per window ----------------
// LDS map (floats):
//   phase A: [0,10800) qkv f32 [9][1200]; [10800,14400) o f32 [9][400]; [14400,14562) scores
//   phase B+: [0,3600) h0; [3600,7200) h1; [7200,14400) ff as f16 [9][1600]
__global__ void __launch_bounds__(256, 1)
window_kernel(const float* __restrict__ xs,
              const float* __restrict__ qkv_all,
              const _Float16* __restrict__ w_out_h,
              const float* __restrict__ b_out,
              const _Float16* __restrict__ w1_h,
              const float* __restrict__ b1,
              const _Float16* __restrict__ w2_h,
              const float* __restrict__ b2,
              const float* __restrict__ ln1_g, const float* __restrict__ ln1_b,
              const float* __restrict__ ln2_g, const float* __restrict__ ln2_b,
              float* __restrict__ out_win) {
  __shared__ float smem[14656];
  int w = blockIdx.x;
  int n = w / W_CNT;
  int i = w % W_CNT;
  int tid = threadIdx.x, wv = tid >> 5, lane = tid & 31;

  // --- gather qkv rows (contiguous, vectorized) ---
  {
    const float4* src = (const float4*)(qkv_all + (size_t)(n * T_LEN + i) * 1200);
    float4* dst = (float4*)smem;
    for (int v4 = tid; v4 < (L_W * 1200) / 4; v4 += 256) dst[v4] = src[v4];
  }
  __syncthreads();

  // --- attention scores + softmax (tiny, VALU) ---
  float* scores = smem + 14400;  // [2][9][9]
  for (int s = tid; s < H_N * L_W * L_W; s += 256) {
    int h = s / 81, r = s % 81, a = r / 9, b = r % 9;
    const float* qp = smem + a * 1200 + h * DH;          // q
    const float* kp = smem + 400 + b * 1200 + h * DH;    // k
    float acc = 0.f;
    for (int e = 0; e < DH; ++e) acc += qp[e] * kp[e];
    scores[s] = acc * 0.07071067811865475f;              // 1/sqrt(200)
  }
  __syncthreads();
  if (tid < H_N * L_W) {
    float* row = scores + tid * L_W;
    float mx = row[0];
#pragma unroll
    for (int b = 1; b < L_W; ++b) mx = fmaxf(mx, row[b]);
    float sum = 0.f;
#pragma unroll
    for (int b = 0; b < L_W; ++b) { row[b] = __expf(row[b] - mx); sum += row[b]; }
    float inv = 1.0f / sum;
#pragma unroll
    for (int b = 0; b < L_W; ++b) row[b] *= inv;
  }
  __syncthreads();

  // --- o = att @ v  -> [9][400] at smem+10800 ---
  float* o = smem + 10800;
  for (int s = tid; s < L_W * D_M; s += 256) {
    int a = s / D_M, dim = s % D_M, h = dim / DH, e = dim % DH;
    const float* att = scores + h * 81 + a * 9;
    float acc = 0.f;
#pragma unroll
    for (int b = 0; b < L_W; ++b) acc += att[b] * smem[800 + b * 1200 + h * DH + e];
    o[s] = acc;
  }
  __syncthreads();

  // --- load h0 (qkv region now dead) ---
  {
    const float4* src = (const float4*)(xs + (size_t)(n * T_LEN + i) * D_M);
    float4* dst = (float4*)smem;
    for (int v4 = tid; v4 < (L_W * D_M) / 4; v4 += 256) dst[v4] = src[v4];
  }
  __syncthreads();
  float* h0 = smem;
  float* h1 = smem + 3600;

  // --- out projection (WMMA, A hoisted, B pipelined) + residual -> h1 ---
  {
    v16h a13[KS_D];
#pragma unroll
    for (int ks = 0; ks < KS_D; ++ks)
      a13[ks] = load_a_f32(o, D_M, L_W, ks * 32, D_M, lane);
    for (int nt = wv; nt < D_M / 16; nt += 8) {
      v8f acc = gemm_pipelined<KS_D>(a13, w_out_h, D_M, nt * 16, lane);
      int half = lane >> 4, nn = nt * 16 + (lane & 15);
      float bo = b_out[nn];
#pragma unroll
      for (int g = 0; g < 8; ++g) {
        int m = g + half * 8;
        if (m < L_W) h1[m * D_M + nn] = acc[g] + bo + h0[m * D_M + nn];
      }
    }
  }
  __syncthreads();
  layernorm_rows(h1, ln1_g, ln1_b, wv, lane);
  __syncthreads();

  // --- FFN1 (WMMA, A hoisted across ~13 tiles/wave) + exact GELU -> ff (f16) ---
  _Float16* ff = (_Float16*)(smem + 7200);  // [9][1600]
  {
    v16h a13[KS_D];
#pragma unroll
    for (int ks = 0; ks < KS_D; ++ks)
      a13[ks] = load_a_f32(h1, D_M, L_W, ks * 32, D_M, lane);
    for (int nt = wv; nt < FF_M / 16; nt += 8) {
      v8f acc = gemm_pipelined<KS_D>(a13, w1_h, D_M, nt * 16, lane);
      int half = lane >> 4, nn = nt * 16 + (lane & 15);
      float bb = b1[nn];
#pragma unroll
      for (int g = 0; g < 8; ++g) {
        int m = g + half * 8;
        if (m < L_W) {
          float xv = acc[g] + bb;
          float gl = 0.5f * xv * (1.0f + erff(xv * 0.7071067811865475f));
          ff[m * FF_M + nn] = (_Float16)gl;
        }
      }
    }
  }
  __syncthreads();

  // --- FFN2 (WMMA, K=1600, K-chunked A hoist, 4 live accumulators/wave) ---
  {
    v8f acc2[4] = {v8f{}, v8f{}, v8f{}, v8f{}};
    for (int kc = 0; kc < 5; ++kc) {          // 5 chunks x 10 ksteps = 50
      v16h a10[10];
#pragma unroll
      for (int ks = 0; ks < 10; ++ks)
        a10[ks] = load_a_f16(ff, FF_M, L_W, kc * 320 + ks * 32, FF_M, lane);
#pragma unroll
      for (int tt = 0; tt < 4; ++tt) {
        int nt = wv + tt * 8;                 // wave-uniform
        if (nt < D_M / 16) {
          v16h bc = load_b_w(w2_h, FF_M, nt * 16, kc * 320, lane);
#pragma unroll
          for (int ks = 0; ks < 10; ++ks) {
            v16h bn = bc;
            if (ks + 1 < 10)
              bn = load_b_w(w2_h, FF_M, nt * 16, kc * 320 + (ks + 1) * 32, lane);
            acc2[tt] = wmma_acc(a10[ks], bc, acc2[tt]);
            bc = bn;
          }
        }
      }
    }
#pragma unroll
    for (int tt = 0; tt < 4; ++tt) {
      int nt = wv + tt * 8;
      if (nt < D_M / 16) {
        int half = lane >> 4, nn = nt * 16 + (lane & 15);
        float bb = b2[nn];
#pragma unroll
        for (int g = 0; g < 8; ++g) {
          int m = g + half * 8;
          if (m < L_W) h1[m * D_M + nn] = h1[m * D_M + nn] + acc2[tt][g] + bb;
        }
      }
    }
  }
  __syncthreads();
  layernorm_rows(h1, ln2_g, ln2_b, wv, lane);
  __syncthreads();

  // --- store window result ---
  {
    const float4* src = (const float4*)h1;
    float4* dst = (float4*)(out_win + (size_t)w * (L_W * D_M));
    for (int v4 = tid; v4 < (L_W * D_M) / 4; v4 += 256) dst[v4] = src[v4];
  }
}

// ---------------- Kernel 5: deterministic overlap-gather + output transpose ----------------
// row t receives out_win[i][t - i/2] for i in [2(t-8), 2t+1] ∩ [0, W)
__global__ void gather_out_kernel(const float* __restrict__ out_win,
                                  float* __restrict__ out) {
  int idx = blockIdx.x * blockDim.x + threadIdx.x;
  if (idx >= NROWS * D_M) return;
  int n = idx / (T_LEN * D_M);
  int r = idx % (T_LEN * D_M);
  int t = r / D_M;
  int d = r % D_M;
  int ilo = 2 * (t - 8); if (ilo < 0) ilo = 0;
  int ihi = 2 * t + 1;   if (ihi > W_CNT - 1) ihi = W_CNT - 1;
  float sum = 0.f;
  for (int i = ilo; i <= ihi; ++i) {
    int j = t - (i >> 1);
    if (j >= 0 && j < L_W)
      sum += out_win[((size_t)(n * W_CNT + i) * L_W + j) * D_M + d];
  }
  int v = d >> 4, c = d & 15;
  out[((n * C_CH + c) * T_LEN + t) * V_J + v] = sum;
}

// ---------------- host launch ----------------
extern "C" void kernel_launch(void* const* d_in, const int* in_sizes, int n_in,
                              void* d_out, int out_size, void* d_ws, size_t ws_size,
                              hipStream_t stream) {
  const float* x     = (const float*)d_in[0];
  const float* pe    = (const float*)d_in[1];
  const float* w_in  = (const float*)d_in[2];
  const float* b_in  = (const float*)d_in[3];
  const float* w_out = (const float*)d_in[4];
  const float* b_out = (const float*)d_in[5];
  const float* w1    = (const float*)d_in[6];
  const float* b1    = (const float*)d_in[7];
  const float* w2    = (const float*)d_in[8];
  const float* b2    = (const float*)d_in[9];
  const float* ln1g  = (const float*)d_in[10];
  const float* ln1b  = (const float*)d_in[11];
  const float* ln2g  = (const float*)d_in[12];
  const float* ln2b  = (const float*)d_in[13];
  float* out = (float*)d_out;

  // workspace layout
  char* ws = (char*)d_ws;
  size_t off = 0;
  float*    xs      = (float*)(ws + off);    off += (size_t)NROWS * D_M * 4;       // 7.68 MB
  _Float16* xs_h    = (_Float16*)(ws + off); off += (size_t)NROWS * D_M * 2;       // 3.84 MB
  _Float16* w_in_h  = (_Float16*)(ws + off); off += (size_t)3 * D_M * D_M * 2;     // 0.96 MB
  _Float16* w_out_h = (_Float16*)(ws + off); off += (size_t)D_M * D_M * 2;         // 0.32 MB
  _Float16* w1_h    = (_Float16*)(ws + off); off += (size_t)FF_M * D_M * 2;        // 1.28 MB
  _Float16* w2_h    = (_Float16*)(ws + off); off += (size_t)D_M * FF_M * 2;        // 1.28 MB
  float*    qkv_all = (float*)(ws + off);    off += (size_t)NROWS * 3 * D_M * 4;   // 23.04 MB
  float*    out_win = (float*)(ws + off);    off += (size_t)NWIN * L_W * D_M * 4;  // 67.05 MB

  const int TPB = 256;

  // 1) layout + positional encoding
  prep_xs_kernel<<<(NROWS * D_M + TPB - 1) / TPB, TPB, 0, stream>>>(x, pe, xs, xs_h);

  // 2) weight conversion to f16
  cvt_f16_kernel<<<(3 * D_M * D_M + TPB - 1) / TPB, TPB, 0, stream>>>(w_in, w_in_h,
                                                                      3 * D_M * D_M);
  cvt_f16_kernel<<<(D_M * D_M + TPB - 1) / TPB, TPB, 0, stream>>>(w_out, w_out_h,
                                                                  D_M * D_M);
  cvt_f16_kernel<<<(FF_M * D_M + TPB - 1) / TPB, TPB, 0, stream>>>(w1, w1_h, FF_M * D_M);
  cvt_f16_kernel<<<(D_M * FF_M + TPB - 1) / TPB, TPB, 0, stream>>>(w2, w2_h, D_M * FF_M);

  // 3) hoisted QKV GEMM: one block per 16-row tile, A fragments register-resident
  qkv_gemm_kernel<<<NROWS / 16, TPB, 0, stream>>>(xs_h, w_in_h, b_in, qkv_all);

  // 4) per-window attention + FFN (WMMA)
  window_kernel<<<NWIN, TPB, 0, stream>>>(xs, qkv_all, w_out_h, b_out, w1_h, b1, w2_h,
                                          b2, ln1g, ln1b, ln2g, ln2b, out_win);

  // 5) deterministic overlap-add + transpose to [N,C,T,V]
  gather_out_kernel<<<(NROWS * D_M + TPB - 1) / TPB, TPB, 0, stream>>>(out_win, out);
}
